// FeatureAlignmentModule_88029649699594
// MI455X (gfx1250) — matrix operand order repaired and checked by reading, not compiled, not run on hardware
//
#include <hip/hip_runtime.h>
#include <hip/hip_bf16.h>

// ---------------------------------------------------------------------------
// FeatureAlignmentModule for MI455X (gfx1250), bf16 WMMA pipeline, v3.
//  - GEMM B operands stored K-major ([P][K]): WMMA fragments are direct
//    16B-aligned global_load_b128 pairs, no LDS staging.
//  - Wave tile 32Mx64N: 8 wmma per K-step vs 12 b128 loads (1.5 loads/wmma).
//  - Output mode templated (f32 vs bf16) -> branch-free epilogue.
//  - K-order for 3x3 GEMMs is (k*256+c) so producers write 8 consecutive c
//    as one b128 store; weights reordered once to match.
//  - BN folded exactly into the 1x1 conv; f32 for stats/offsets/bilinear.
// ---------------------------------------------------------------------------

typedef __bf16 bf16;
typedef __attribute__((ext_vector_type(16))) __bf16 v16bf;
typedef __attribute__((ext_vector_type(8)))  __bf16 v8bf;
typedef __attribute__((ext_vector_type(8)))  float  v8f;

constexpr int Cch = 256;
constexpr int HWp = 4096;              // 64*64
constexpr int CHW = Cch * HWp;         // per-batch plane stride
constexpr int NB  = 4;
constexpr int KD3 = 9 * Cch;           // 2304, K of the 3x3 GEMMs

static __device__ __forceinline__ int imin(int a, int b) { return a < b ? a : b; }
static __device__ __forceinline__ int imax(int a, int b) { return a > b ? a : b; }

// ---------------- BN statistics: one block per channel --------------------
__global__ __launch_bounds__(256)
void k_bn_stats(const float* __restrict__ x, float* __restrict__ mean,
                float* __restrict__ rstd) {
    __shared__ float s1[256], s2[256];
    int c = blockIdx.x, t = threadIdx.x;
    float s = 0.f, sq = 0.f;
    for (int b = 0; b < NB; ++b) {
        const float* p = x + ((size_t)b * Cch + c) * HWp;
        for (int i = t; i < HWp; i += 256) { float v = p[i]; s += v; sq += v * v; }
    }
    s1[t] = s; s2[t] = sq; __syncthreads();
    for (int o = 128; o > 0; o >>= 1) {
        if (t < o) { s1[t] += s1[t + o]; s2[t] += s2[t + o]; }
        __syncthreads();
    }
    if (t == 0) {
        float m = s1[0] * (1.f / 16384.f);
        float v = s2[0] * (1.f / 16384.f) - m * m;
        mean[c] = m;
        rstd[c] = rsqrtf(v + 1e-5f);
    }
}

// ------------- fold BN into 1x1 conv: w1eff (bf16), b1eff -----------------
__global__ __launch_bounds__(256)
void k_fold(const float* __restrict__ w1, const float* __restrict__ b1,
            const float* __restrict__ gamma, const float* __restrict__ beta,
            const float* __restrict__ mean, const float* __restrict__ rstd,
            bf16* __restrict__ w1eff, float* __restrict__ b1eff) {
    __shared__ float red[256];
    int o = blockIdx.x, c = threadIdx.x;
    float s = gamma[c] * rstd[c];
    float w = w1[o * Cch + c];
    w1eff[o * Cch + c] = (bf16)(w * s);
    float t = beta[c] - mean[c] * s;
    red[c] = w * t; __syncthreads();
    for (int k = 128; k > 0; k >>= 1) {
        if (c < k) red[c] += red[c + k];
        __syncthreads();
    }
    if (c == 0) b1eff[o] = b1[o] + red[0];
}

// ------- transpose x NCHW(f32) -> xbfT [b][p][c](bf16), padded LDS --------
__global__ __launch_bounds__(256)
void k_xpose(const float* __restrict__ x, bf16* __restrict__ xt) {
    __shared__ bf16 t[32][33];
    int bx = blockIdx.x * 32;                // p base
    int by = blockIdx.y * 32;                // c base
    int b  = blockIdx.z;
    int tx = threadIdx.x & 31, ty = threadIdx.x >> 5;
    const float* src = x + (size_t)b * CHW;
    #pragma unroll
    for (int i = 0; i < 4; ++i) {
        int c = by + ty + i * 8;
        t[ty + i * 8][tx] = (bf16)src[(size_t)c * HWp + bx + tx];
    }
    __syncthreads();
    bf16* dst = xt + (size_t)b * CHW;
    #pragma unroll
    for (int i = 0; i < 4; ++i) {
        int p = bx + ty + i * 8;
        dst[(size_t)p * Cch + by + tx] = t[tx][ty + i * 8];
    }
}

// ---- reorder conv weights [O][C][9](f32) -> [O][k*256+c](bf16) ----------
__global__ __launch_bounds__(256)
void k_wxpose(const float* __restrict__ w, bf16* __restrict__ wt, int n) {
    int i = blockIdx.x * 256 + threadIdx.x;
    if (i >= n) return;
    int o = i / KD3, r = i % KD3;
    int k = r >> 8, c = r & 255;
    wt[i] = (bf16)w[((size_t)o * Cch + c) * 9 + k];
}

// ------------------------------- WMMA GEMM --------------------------------
// D[256 x 4096] = A[256 x KD](bf16 row-major) * Bt[4096 x KD](bf16, K-major)
// block: 256 thr = 8 waves (2M x 4N); wave tile 32Mx64N; block 64Mx256N.
// Per K-step: 4 b128 (two A frags) + 8 b128 (four B frags) -> 8 wmma.
template <int KD, bool OUTF>
__global__ __launch_bounds__(256)
void k_gemm(const bf16* __restrict__ A, const bf16* __restrict__ Bt,
            const float* __restrict__ bias, float* __restrict__ outF,
            bf16* __restrict__ outB) {
    int tid  = threadIdx.x;
    int lane = tid & 31;
    int wid  = tid >> 5;
    int lh   = lane >> 4;                      // lane half 0/1
    int ll   = lane & 15;
    int m0 = blockIdx.x * 64 + (wid & 1) * 32; // wave M base (32 rows)
    int n0 = blockIdx.y * 256 + (wid >> 1) * 64;

    v8f acc[2][4] = {};

    const unsigned int* ap0 =
        (const unsigned int*)A + (size_t)(m0 + ll) * (KD >> 1) + (lh << 3);
    const unsigned int* ap1 = ap0 + (size_t)16 * (KD >> 1);
    const bf16* bp0 = Bt + (size_t)(n0 + 0 * 16 + ll) * KD + (lh << 4);
    const bf16* bp1 = Bt + (size_t)(n0 + 1 * 16 + ll) * KD + (lh << 4);
    const bf16* bp2 = Bt + (size_t)(n0 + 2 * 16 + ll) * KD + (lh << 4);
    const bf16* bp3 = Bt + (size_t)(n0 + 3 * 16 + ll) * KD + (lh << 4);

    #pragma unroll 2
    for (int k0 = 0; k0 < KD; k0 += 32) {
        // A fragments, contiguous K: lane half h holds K 16h..16h+15
        union { v16bf v; unsigned int u[8]; } af0, af1;
        const unsigned int* a0 = ap0 + (k0 >> 1);
        const unsigned int* a1 = ap1 + (k0 >> 1);
        af0.u[0] = a0[0]; af0.u[1] = a0[1]; af0.u[2] = a0[2];  af0.u[3] = a0[3];
        af0.u[4] = a0[4]; af0.u[5] = a0[5]; af0.u[6] = a0[6]; af0.u[7] = a0[7];
        af1.u[0] = a1[0]; af1.u[1] = a1[1]; af1.u[2] = a1[2];  af1.u[3] = a1[3];
        af1.u[4] = a1[4]; af1.u[5] = a1[5]; af1.u[6] = a1[6]; af1.u[7] = a1[7];

        union { v16bf v; v8bf h[2]; } b0, b1, b2, b3;
        b0.h[0] = *(const v8bf*)(bp0 + k0); b0.h[1] = *(const v8bf*)(bp0 + k0 + 8);
        b1.h[0] = *(const v8bf*)(bp1 + k0); b1.h[1] = *(const v8bf*)(bp1 + k0 + 8);
        b2.h[0] = *(const v8bf*)(bp2 + k0); b2.h[1] = *(const v8bf*)(bp2 + k0 + 8);
        b3.h[0] = *(const v8bf*)(bp3 + k0); b3.h[1] = *(const v8bf*)(bp3 + k0 + 8);

        acc[0][0] = __builtin_amdgcn_wmma_f32_16x16x32_bf16(
            false, af0.v, false, b0.v, (short)0, acc[0][0], false, false);
        acc[0][1] = __builtin_amdgcn_wmma_f32_16x16x32_bf16(
            false, af0.v, false, b1.v, (short)0, acc[0][1], false, false);
        acc[0][2] = __builtin_amdgcn_wmma_f32_16x16x32_bf16(
            false, af0.v, false, b2.v, (short)0, acc[0][2], false, false);
        acc[0][3] = __builtin_amdgcn_wmma_f32_16x16x32_bf16(
            false, af0.v, false, b3.v, (short)0, acc[0][3], false, false);
        acc[1][0] = __builtin_amdgcn_wmma_f32_16x16x32_bf16(
            false, af1.v, false, b0.v, (short)0, acc[1][0], false, false);
        acc[1][1] = __builtin_amdgcn_wmma_f32_16x16x32_bf16(
            false, af1.v, false, b1.v, (short)0, acc[1][1], false, false);
        acc[1][2] = __builtin_amdgcn_wmma_f32_16x16x32_bf16(
            false, af1.v, false, b2.v, (short)0, acc[1][2], false, false);
        acc[1][3] = __builtin_amdgcn_wmma_f32_16x16x32_bf16(
            false, af1.v, false, b3.v, (short)0, acc[1][3], false, false);
    }
    // write-back per f32 C/D layout: VGPR v -> M = mbase + 8*lh + v
    #pragma unroll
    for (int i = 0; i < 2; ++i) {
        int mb = m0 + i * 16 + (lh << 3);
        #pragma unroll
        for (int f = 0; f < 4; ++f) {
            int n = n0 + f * 16 + ll;
            #pragma unroll
            for (int v = 0; v < 8; ++v) {
                int m = mb + v;
                float val = acc[i][f][v] + bias[m];
                if (OUTF) outF[(size_t)m * HWp + n] = val;
                else      outB[(size_t)m * HWp + n] = (bf16)val;
            }
        }
    }
}

// ------------------ offset conv: 3x3, 256 -> 18 ch, f32 -------------------
__global__ __launch_bounds__(256)
void k_off_conv(const float* __restrict__ y1, const float* __restrict__ woff,
                const float* __restrict__ boff, float* __restrict__ off) {
    __shared__ float wsh[18 * 32 * 9];
    int tid = threadIdx.x;
    int b = blockIdx.y;
    int p = blockIdx.x * 256 + tid;
    int h = p >> 6, w = p & 63;
    float acc[18];
    for (int o = 0; o < 18; ++o) acc[o] = 0.f;
    for (int c0 = 0; c0 < Cch; c0 += 32) {
        __syncthreads();
        for (int i = tid; i < 18 * 32 * 9; i += 256) {
            int o = i / 288, r = i % 288, cc = r / 9, tap = r % 9;
            wsh[i] = woff[(size_t)(o * Cch + c0 + cc) * 9 + tap];
        }
        __syncthreads();
        for (int cc = 0; cc < 32; ++cc) {
            const float* img = y1 + ((size_t)b * Cch + c0 + cc) * HWp;
            for (int tap = 0; tap < 9; ++tap) {
                int hh = h + tap / 3 - 1, ww = w + tap % 3 - 1;
                float v = 0.f;
                if (hh >= 0 && hh < 64 && ww >= 0 && ww < 64) v = img[hh * 64 + ww];
                const float* wrow = wsh + cc * 9 + tap;
                for (int o = 0; o < 18; ++o) acc[o] += v * wrow[o * 288];
            }
        }
    }
    for (int o = 0; o < 18; ++o)
        off[((size_t)b * 18 + o) * HWp + p] = acc[o] + boff[o];
}

// ----- deform bilinear sample -> colT[p][k*256+c](bf16), b128 stores ------
__global__ __launch_bounds__(256)
void k_deform(const float* __restrict__ y1b, const float* __restrict__ offb,
              bf16* __restrict__ colT) {
    int tid = threadIdx.x;
    int p = blockIdx.x * 256 + tid;
    int k = blockIdx.y;                        // tap 0..8
    int h = p >> 6, w = p & 63;
    float oy = offb[(size_t)(2 * k) * HWp + p];
    float ox = offb[(size_t)(2 * k + 1) * HWp + p];
    float py = oy + (float)(h - 1 + k / 3);
    float px = ox + (float)(w - 1 + k % 3);
    float y0f = floorf(py), x0f = floorf(px);
    float wy = py - y0f, wx = px - x0f;
    int y0 = (int)y0f, x0 = (int)x0f;
    int y1i = y0 + 1, x1i = x0 + 1;
    bool vy0 = (y0 >= 0) && (y0 < 64);
    bool vy1 = (y1i >= 0) && (y1i < 64);
    bool vx0 = (x0 >= 0) && (x0 < 64);
    bool vx1 = (x1i >= 0) && (x1i < 64);
    int cy0 = imin(imax(y0, 0), 63),  cy1 = imin(imax(y1i, 0), 63);
    int cx0 = imin(imax(x0, 0), 63),  cx1 = imin(imax(x1i, 0), 63);
    int i00 = cy0 * 64 + cx0, i01 = cy0 * 64 + cx1;
    int i10 = cy1 * 64 + cx0, i11 = cy1 * 64 + cx1;
    float w00 = (vy0 && vx0) ? (1.f - wy) * (1.f - wx) : 0.f;
    float w01 = (vy0 && vx1) ? (1.f - wy) * wx : 0.f;
    float w10 = (vy1 && vx0) ? wy * (1.f - wx) : 0.f;
    float w11 = (vy1 && vx1) ? wy * wx : 0.f;
    bf16* dst = colT + (size_t)p * KD3 + k * 256;
    for (int c0 = 0; c0 < Cch; c0 += 8) {
        v8bf vv;
        #pragma unroll
        for (int j = 0; j < 8; ++j) {
            const float* img = y1b + (size_t)(c0 + j) * HWp;
            float v = w00 * img[i00] + w01 * img[i01] +
                      w10 * img[i10] + w11 * img[i11];
            vv[j] = (bf16)v;
        }
        *(v8bf*)(dst + c0) = vv;                 // 16B-aligned b128 store
    }
}

// -------- plain 3x3 im2col: y2bf[c][p] -> colT[p][k*256+c](bf16) ----------
__global__ __launch_bounds__(256)
void k_im2col(const bf16* __restrict__ y2b, bf16* __restrict__ colT) {
    int tid = threadIdx.x;
    int p = blockIdx.x * 256 + tid;
    int k = blockIdx.y;
    int h = p >> 6, w = p & 63;
    int hh = h + k / 3 - 1, ww = w + k % 3 - 1;
    bool valid = (hh >= 0) && (hh < 64) && (ww >= 0) && (ww < 64);
    int ps = hh * 64 + ww;
    bf16* dst = colT + (size_t)p * KD3 + k * 256;
    for (int c0 = 0; c0 < Cch; c0 += 8) {
        v8bf vv;
        #pragma unroll
        for (int j = 0; j < 8; ++j)
            vv[j] = valid ? y2b[(size_t)(c0 + j) * HWp + ps] : (bf16)0.f;
        *(v8bf*)(dst + c0) = vv;
    }
}

// ---------------------------------------------------------------------------
static inline void* bump(char*& p, size_t bytes) {
    void* r = (void*)p;
    p += (bytes + 255) & ~(size_t)255;
    return r;
}

extern "C" void kernel_launch(void* const* d_in, const int* in_sizes, int n_in,
                              void* d_out, int out_size, void* d_ws, size_t ws_size,
                              hipStream_t stream) {
    (void)in_sizes; (void)n_in; (void)out_size; (void)ws_size;

    const float* x     = (const float*)d_in[0];
    const float* gamma = (const float*)d_in[1];
    const float* beta  = (const float*)d_in[2];
    const float* w1    = (const float*)d_in[3];
    const float* b1    = (const float*)d_in[4];
    const float* w_off = (const float*)d_in[5];
    const float* b_off = (const float*)d_in[6];
    const float* w_def = (const float*)d_in[7];
    const float* b_def = (const float*)d_in[8];
    const float* w3    = (const float*)d_in[9];
    const float* b3    = (const float*)d_in[10];
    float* out = (float*)d_out;

    // workspace carve-out (~50 MB total)
    char* ws = (char*)d_ws;
    float* mean   = (float*)bump(ws, Cch * 4);
    float* rstd   = (float*)bump(ws, Cch * 4);
    bf16*  w1eff  = (bf16*) bump(ws, (size_t)Cch * Cch * 2);
    float* b1eff  = (float*)bump(ws, Cch * 4);
    bf16*  xbfT   = (bf16*) bump(ws, (size_t)NB * CHW * 2);          // 8.4 MB [b][p][c]
    float* y1     = (float*)bump(ws, (size_t)NB * CHW * 4);          // 16.8 MB [b][c][p]
    float* offbuf = (float*)bump(ws, (size_t)NB * 18 * HWp * 4);     // 1.2 MB
    bf16*  wdef2  = (bf16*) bump(ws, (size_t)Cch * KD3 * 2);         // 1.2 MB [o][k*256+c]
    bf16*  w32    = (bf16*) bump(ws, (size_t)Cch * KD3 * 2);         // 1.2 MB
    bf16*  colT   = (bf16*) bump(ws, (size_t)HWp * KD3 * 2);         // 18.9 MB [p][kd] (reused)
    bf16*  y2bf   = (bf16*) bump(ws, (size_t)Cch * HWp * 2);         // 2.1 MB [c][p]

    // 1) BN statistics, fold BN into 1x1 conv weights
    k_bn_stats<<<Cch, 256, 0, stream>>>(x, mean, rstd);
    k_fold<<<Cch, 256, 0, stream>>>(w1, b1, gamma, beta, mean, rstd, w1eff, b1eff);

    // 2) layout conversions: x -> [p][c] bf16; 3x3 weights -> [o][k*256+c] bf16
    k_xpose<<<dim3(HWp / 32, Cch / 32, NB), 256, 0, stream>>>(x, xbfT);
    k_wxpose<<<(Cch * KD3 + 255) / 256, 256, 0, stream>>>(w_def, wdef2, Cch * KD3);
    k_wxpose<<<(Cch * KD3 + 255) / 256, 256, 0, stream>>>(w3, w32, Cch * KD3);

    // 3) GEMM1: y1 = (BN-folded 1x1 conv)(x), f32 output
    for (int b = 0; b < NB; ++b)
        k_gemm<Cch, true><<<dim3(4, 16), 256, 0, stream>>>(
            w1eff, xbfT + (size_t)b * CHW, b1eff,
            y1 + (size_t)b * CHW, (bf16*)nullptr);

    // 4) offset conv (f32, 18 ch)
    k_off_conv<<<dim3(16, NB), 256, 0, stream>>>(y1, w_off, b_off, offbuf);

    // 5) per-batch: deform sample -> GEMM2 -> im2col -> GEMM3
    for (int b = 0; b < NB; ++b) {
        k_deform<<<dim3(16, 9), 256, 0, stream>>>(
            y1 + (size_t)b * CHW, offbuf + (size_t)b * 18 * HWp, colT);
        k_gemm<KD3, false><<<dim3(4, 16), 256, 0, stream>>>(
            wdef2, colT, b_def, (float*)nullptr, y2bf);
        k_im2col<<<dim3(16, 9), 256, 0, stream>>>(y2bf, colT);
        k_gemm<KD3, true><<<dim3(4, 16), 256, 0, stream>>>(
            w32, colT, b3, out + (size_t)b * CHW, (bf16*)nullptr);
    }
}
